// MoEMLP_5196910428724
// MI455X (gfx1250) — matrix operand order repaired and checked by reading, not compiled, not run on hardware
//
#include <hip/hip_runtime.h>
#include <hip/hip_bf16.h>
#include <math.h>

// ---------------------------------------------------------------------------
// MoE MLP (B=2,S=2048,H=1024,I=4096,E=8), every expert sees every token,
// expert e weighted by the e-th LARGEST softmax prob (sorted probs).
//
// MI455X / gfx1250 strategy:
//   * 550 GFLOP vs ~300MB traffic -> compute-bound; bf16 WMMA (k=32) path.
//   * per expert: GEMM1 (X @ W1^T, exact GELU, bf16 -> ws Hmid)
//                 GEMM2 (Hmid @ W2^T, scale by sorted prob, accum fp32 out)
//   * 128x128x32 tiles, 8 waves/block (4x2), each wave 2x4 WMMA 16x16 tiles.
//   * global->LDS staging: GLOBAL_LOAD_ASYNC_TO_LDS_B128 (ASYNCcnt), GVS
//     addressing (uniform SGPR base advances with k; 32-bit per-thread
//     voffset), double-buffered LDS, counted s_wait_asynccnt 0x4 overlap.
//     Last k-tile peeled so the hot loop has no conditional branches.
//
// Workspace layout (bytes), total ~176.3 MB:
//   Xb   @ 0          : N*H bf16        =   8,388,608
//   W1b  @ 8,388,608  : E*I*H bf16     =  67,108,864
//   W2b  @ 75,497,472 : E*H*I bf16     =  67,108,864
//   Hb   @142,606,336 : N*I bf16       =  33,554,432   (reused per expert)
//   P    @176,160,768 : N*E f32        =     131,072
// ---------------------------------------------------------------------------

typedef __bf16 v16bf __attribute__((ext_vector_type(16)));
typedef float  v8f   __attribute__((ext_vector_type(8)));

static constexpr int Nn = 4096;   // B*S tokens
static constexpr int Hn = 1024;
static constexpr int In = 4096;
static constexpr int En = 8;

static constexpr size_t WS_XB  = 0;
static constexpr size_t WS_W1B = WS_XB  + (size_t)Nn * Hn * 2;
static constexpr size_t WS_W2B = WS_W1B + (size_t)En * In * Hn * 2;
static constexpr size_t WS_HB  = WS_W2B + (size_t)En * Hn * In * 2;
static constexpr size_t WS_P   = WS_HB  + (size_t)Nn * In * 2;

union FragU { v16bf v; uint4 q[2]; };

__device__ __forceinline__ ushort f2bf(float f) {
    unsigned u = __float_as_uint(f);
    u = (u + 0x7FFFu + ((u >> 16) & 1u)) >> 16;   // round-to-nearest-even
    return (ushort)u;
}

__device__ __forceinline__ float gelu_exact(float x) {
    return 0.5f * x * (1.0f + erff(x * 0.70710678118654752f));
}

// Async 16B copy global->LDS, GVS addressing: addr = SGPR64 base + VGPR32 off.
// Tracked by ASYNCcnt; writes LDS directly (no staging VGPRs).
__device__ __forceinline__ void async_copy_b128_gvs(unsigned lds_off,
                                                    unsigned voff,
                                                    unsigned long long sbase) {
    asm volatile("global_load_async_to_lds_b128 %0, %1, %2"
                 :: "v"(lds_off), "v"(voff), "s"(sbase)
                 : "memory");
}

// ------------------------------- converters --------------------------------
__global__ void f32_to_bf16_kernel(const float* __restrict__ src,
                                   ushort* __restrict__ dst, size_t n) {
    size_t i = ((size_t)blockIdx.x * blockDim.x + threadIdx.x) * 4;
    size_t stride = (size_t)gridDim.x * blockDim.x * 4;
    for (; i < n; i += stride) {
        float4 f = *(const float4*)(src + i);
        uint2 p;
        p.x = (unsigned)f2bf(f.x) | ((unsigned)f2bf(f.y) << 16);
        p.y = (unsigned)f2bf(f.z) | ((unsigned)f2bf(f.w) << 16);
        *(uint2*)(dst + i) = p;
    }
}

// ------------------------------- router ------------------------------------
// One wave32 per token: logits = x . Wr[e], softmax over 8, sort descending.
__global__ void router_kernel(const float* __restrict__ X,
                              const float* __restrict__ Wr,
                              float* __restrict__ P) {
    const int lane  = threadIdx.x & 31;
    const int wave  = threadIdx.x >> 5;
    const int token = blockIdx.x * 8 + wave;
    const float* xrow = X + (size_t)token * Hn;

    float logits[En];
#pragma unroll
    for (int e = 0; e < En; ++e) {
        const float* wrow = Wr + (size_t)e * Hn;
        float s = 0.f;
        for (int h = lane; h < Hn; h += 32) s += xrow[h] * wrow[h];
#pragma unroll
        for (int off = 16; off > 0; off >>= 1) s += __shfl_xor(s, off, 32);
        logits[e] = s;   // every lane has the full sum (butterfly)
    }
    if (lane == 0) {
        float mx = logits[0];
#pragma unroll
        for (int e = 1; e < En; ++e) mx = fmaxf(mx, logits[e]);
        float sum = 0.f, p[En];
#pragma unroll
        for (int e = 0; e < En; ++e) { p[e] = __expf(logits[e] - mx); sum += p[e]; }
        float inv = 1.0f / sum;
#pragma unroll
        for (int e = 0; e < En; ++e) p[e] *= inv;
        // sort descending (8 elements)
#pragma unroll
        for (int i = 0; i < En - 1; ++i)
#pragma unroll
            for (int j = 0; j < En - 1 - i; ++j)
                if (p[j] < p[j + 1]) { float t = p[j]; p[j] = p[j + 1]; p[j + 1] = t; }
#pragma unroll
        for (int e = 0; e < En; ++e) P[(size_t)token * En + e] = p[e];
    }
}

// ------------------------- WMMA GEMM main loop -----------------------------
// C(128x128) += A(128xK) * B^T ; A,B row-major bf16, K-contiguous rows
// (B row n is output column n). 256 threads = 8 waves, wave (m,n) grid 4x2,
// each wave 2x4 tiles of 16x16. Double-buffered LDS filled by async copies;
// LDS row stride padded to 40 ushorts (80B) for conflict-free b128 frag loads.
__device__ __forceinline__ void wmma_tile_mainloop(
    const ushort* __restrict__ Ag, const ushort* __restrict__ Bg,
    int K, int mBase, int nBase, v8f acc[2][4]) {

    constexpr int BK = 32, LDSS = 40;
    constexpr unsigned BUFB = 128 * LDSS * 2;   // bytes per tile buffer
    __shared__ __align__(16) ushort Al[2][128 * LDSS];
    __shared__ __align__(16) ushort Bl[2][128 * LDSS];

    const int tid      = threadIdx.x;
    const int lane     = tid & 31;
    const int wave     = tid >> 5;
    const int waveM    = (wave >> 1) * 32;
    const int waveN    = (wave & 1) * 64;
    const int laneHalf = (lane < 16) ? 0 : 1;
    const int lrow     = lane & 15;

    // Per-thread copy slots: 4 async b128 per tile (2 rows x A,B).
    const int row0 = tid >> 2;            // 0..63
    const int row1 = row0 + 64;           // 64..127
    const int q0   = tid & 3;             // 16B chunk within 64B row

    // Loop-invariant 32-bit byte voffsets (k-step advances the SGPR base).
    const unsigned voffA0 = (unsigned)(((size_t)(mBase + row0) * K + q0 * 8) * 2);
    const unsigned voffA1 = (unsigned)(((size_t)(mBase + row1) * K + q0 * 8) * 2);
    const unsigned voffB0 = (unsigned)(((size_t)(nBase + row0) * K + q0 * 8) * 2);
    const unsigned voffB1 = (unsigned)(((size_t)(nBase + row1) * K + q0 * 8) * 2);
    // LDS byte offsets (low 32 bits of flat shared address = LDS offset).
    const unsigned la0 = (unsigned)(uintptr_t)&Al[0][row0 * LDSS + q0 * 8];
    const unsigned la1 = (unsigned)(uintptr_t)&Al[0][row1 * LDSS + q0 * 8];
    const unsigned lb0 = (unsigned)(uintptr_t)&Bl[0][row0 * LDSS + q0 * 8];
    const unsigned lb1 = (unsigned)(uintptr_t)&Bl[0][row1 * LDSS + q0 * 8];

    const unsigned long long baseA0 = (unsigned long long)(uintptr_t)Ag;
    const unsigned long long baseB0 = (unsigned long long)(uintptr_t)Bg;

    auto issue_tile = [&](int k0, int buf) {
        const unsigned long long ba = baseA0 + (unsigned long long)k0 * 2;
        const unsigned long long bb = baseB0 + (unsigned long long)k0 * 2;
        const unsigned bo = (unsigned)buf * BUFB;
        async_copy_b128_gvs(la0 + bo, voffA0, ba);
        async_copy_b128_gvs(lb0 + bo, voffB0, bb);
        async_copy_b128_gvs(la1 + bo, voffA1, ba);
        async_copy_b128_gvs(lb1 + bo, voffB1, bb);
    };

    // Fragment layouts per ISA 7.12.2:
    //  A 16x32: lanes0-15 rows M, K 0..7 & 16..23; lanes16-31 K 8..15 & 24..31
    //  B 32x16: lanes0-15 col N, K 0..7 & 16..23; lanes16-31 K 8..15 & 24..31
    auto compute_tile = [&](int buf) {
        const ushort* Ab = &Al[buf][0];
        const ushort* Bb = &Bl[buf][0];
        FragU a[2], b[4];
#pragma unroll
        for (int tm = 0; tm < 2; ++tm) {
            const ushort* base = Ab + (waveM + tm * 16 + lrow) * LDSS;
            a[tm].q[0] = *(const uint4*)(base + laneHalf * 8);
            a[tm].q[1] = *(const uint4*)(base + laneHalf * 8 + 16);
        }
#pragma unroll
        for (int tn = 0; tn < 4; ++tn) {
            const ushort* base = Bb + (waveN + tn * 16 + lrow) * LDSS;
            b[tn].q[0] = *(const uint4*)(base + laneHalf * 8);
            b[tn].q[1] = *(const uint4*)(base + laneHalf * 8 + 16);
        }
#pragma unroll
        for (int tm = 0; tm < 2; ++tm)
#pragma unroll
            for (int tn = 0; tn < 4; ++tn)
                acc[tm][tn] = __builtin_amdgcn_wmma_f32_16x16x32_bf16(
                    false, a[tm].v, false, b[tn].v,
                    (short)0, acc[tm][tn], false, false);
    };

    const int ntiles = K / BK;
    issue_tile(0, 0);                                  // prologue

    // Hot loop: branch-free body (last tile peeled).
#pragma unroll 1
    for (int t = 0; t < ntiles - 1; ++t) {
        const int buf = t & 1;
        issue_tile((t + 1) * BK, buf ^ 1);             // overlap with compute
        // 8 in flight; wait until only the 4 newest remain -> tile t done
        asm volatile("s_wait_asynccnt 0x4" ::: "memory");
        __syncthreads();   // all waves' async writes to buf are visible
        compute_tile(buf);
        __syncthreads();   // reads done before next iter writes this buffer
    }

    // Epilogue: last tile.
    asm volatile("s_wait_asynccnt 0x0" ::: "memory");
    __syncthreads();
    compute_tile((ntiles - 1) & 1);
}

// ------------------------- GEMM1: X @ W1^T, GELU ---------------------------
__global__ void __launch_bounds__(256, 1)
gemm1_gelu_kernel(const ushort* __restrict__ Xb,
                  const ushort* __restrict__ W1b,
                  ushort* __restrict__ Hb, int e) {
    const int mBase = blockIdx.x * 128;
    const int nBase = blockIdx.y * 128;
    const ushort* Bg = W1b + (size_t)e * In * Hn;

    v8f acc[2][4];
    const v8f zero = {0.f, 0.f, 0.f, 0.f, 0.f, 0.f, 0.f, 0.f};
#pragma unroll
    for (int tm = 0; tm < 2; ++tm)
#pragma unroll
        for (int tn = 0; tn < 4; ++tn) acc[tm][tn] = zero;

    wmma_tile_mainloop(Xb, Bg, Hn, mBase, nBase, acc);

    const int lane  = threadIdx.x & 31;
    const int wave  = threadIdx.x >> 5;
    const int waveM = (wave >> 1) * 32;
    const int waveN = (wave & 1) * 64;
    const int rAdd  = (lane < 16) ? 0 : 8;   // C layout: VGPR r -> M=r / r+8
    const int col0  = lane & 15;
#pragma unroll
    for (int tm = 0; tm < 2; ++tm)
#pragma unroll
        for (int tn = 0; tn < 4; ++tn)
#pragma unroll
            for (int r = 0; r < 8; ++r) {
                int row = mBase + waveM + tm * 16 + r + rAdd;
                int col = nBase + waveN + tn * 16 + col0;
                Hb[(size_t)row * In + col] = f2bf(gelu_exact(acc[tm][tn][r]));
            }
}

// ---------------- GEMM2: Hmid @ W2^T, prob-weighted combine ----------------
__global__ void __launch_bounds__(256, 1)
gemm2_combine_kernel(const ushort* __restrict__ Hb,
                     const ushort* __restrict__ W2b,
                     const float* __restrict__ P,
                     float* __restrict__ out, int e) {
    const int mBase = blockIdx.x * 128;
    const int nBase = blockIdx.y * 128;
    const ushort* Bg = W2b + (size_t)e * Hn * In;

    v8f acc[2][4];
    const v8f zero = {0.f, 0.f, 0.f, 0.f, 0.f, 0.f, 0.f, 0.f};
#pragma unroll
    for (int tm = 0; tm < 2; ++tm)
#pragma unroll
        for (int tn = 0; tn < 4; ++tn) acc[tm][tn] = zero;

    wmma_tile_mainloop(Hb, Bg, In, mBase, nBase, acc);

    const int lane  = threadIdx.x & 31;
    const int wave  = threadIdx.x >> 5;
    const int waveM = (wave >> 1) * 32;
    const int waveN = (wave & 1) * 64;
    const int rAdd  = (lane < 16) ? 0 : 8;
    const int col0  = lane & 15;
#pragma unroll
    for (int tm = 0; tm < 2; ++tm)
#pragma unroll
        for (int tn = 0; tn < 4; ++tn)
#pragma unroll
            for (int r = 0; r < 8; ++r) {
                int row = mBase + waveM + tm * 16 + r + rAdd;   // token id
                int col = nBase + waveN + tn * 16 + col0;
                float p = P[(size_t)row * En + e];               // e-th largest
                float v = acc[tm][tn][r] * p;
                float* dst = out + (size_t)row * Hn + col;
                if (e == 0) *dst = v;       // first expert initializes
                else        *dst += v;      // later experts accumulate
            }
}

// ------------------------------- launcher ----------------------------------
extern "C" void kernel_launch(void* const* d_in, const int* in_sizes, int n_in,
                              void* d_out, int out_size, void* d_ws, size_t ws_size,
                              hipStream_t stream) {
    (void)in_sizes; (void)n_in; (void)out_size; (void)ws_size;
    const float* X  = (const float*)d_in[0];   // (B,S,H)
    const float* Wr = (const float*)d_in[1];   // (E,H)
    const float* W1 = (const float*)d_in[2];   // (E,I,H)
    const float* W2 = (const float*)d_in[3];   // (E,H,I)
    float* out = (float*)d_out;

    char* ws = (char*)d_ws;
    ushort* Xb  = (ushort*)(ws + WS_XB);
    ushort* W1b = (ushort*)(ws + WS_W1B);
    ushort* W2b = (ushort*)(ws + WS_W2B);
    ushort* Hb  = (ushort*)(ws + WS_HB);
    float*  P   = (float*)(ws + WS_P);

    f32_to_bf16_kernel<<<1024, 256, 0, stream>>>(X, Xb, (size_t)Nn * Hn);
    f32_to_bf16_kernel<<<4096, 256, 0, stream>>>(W1, W1b, (size_t)En * In * Hn);
    f32_to_bf16_kernel<<<4096, 256, 0, stream>>>(W2, W2b, (size_t)En * Hn * In);
    router_kernel<<<Nn / 8, 256, 0, stream>>>(X, Wr, P);

    for (int e = 0; e < En; ++e) {
        gemm1_gelu_kernel<<<dim3(Nn / 128, In / 128), 256, 0, stream>>>(Xb, W1b, Hb, e);
        gemm2_combine_kernel<<<dim3(Nn / 128, Hn / 128), 256, 0, stream>>>(Hb, W2b, P, out, e);
    }
}
